// TransformerBlock_9655086481991
// MI455X (gfx1250) — matrix operand order repaired and checked by reading, not compiled, not run on hardware
//
#include <hip/hip_runtime.h>

// ---------------------------------------------------------------------------
// Transformer block for MI455X (gfx1250, wave32, WMMA).
// B=2, T=2048, C=1024, H=16, Dh=64, FF=4096.
// All GEMMs in bf16 with f32 accumulation via v_wmma_f32_16x16x32_bf16.
// Attention is flash-style (online softmax), per-wave private LDS tiles.
// Workspace requirement: ~120 MB.
// ---------------------------------------------------------------------------

typedef unsigned short u16;
typedef unsigned int   u32;
typedef __attribute__((ext_vector_type(16))) __bf16 v16bf;
typedef __attribute__((ext_vector_type(8)))  float  v8f;

union FragB { uint4 q[2]; v16bf v; };   // 32 bytes = 16 bf16 per lane

__device__ __forceinline__ u16 f2bf(float x) {
  u32 u = __builtin_bit_cast(u32, x);
  return (u16)((u + 0x7FFFu + ((u >> 16) & 1u)) >> 16);
}
__device__ __forceinline__ u32 pk2(float a, float b) {
  return (u32)f2bf(a) | ((u32)f2bf(b) << 16);
}

__device__ __forceinline__ v8f wmma_bf(v16bf a, v16bf b, v8f c) {
  // D = A(16x32 bf16) x B(32x16 bf16) + C(16x16 f32)
  return __builtin_amdgcn_wmma_f32_16x16x32_bf16(false, a, false, b,
                                                 (short)0, c, false, false);
}

__device__ __forceinline__ v8f v8f_zero() {
  v8f z;
#pragma unroll
  for (int i = 0; i < 8; ++i) z[i] = 0.0f;
  return z;
}

// ---------------------------------------------------------------------------
// Weight transpose + f32->bf16 convert:  W[K][N] f32  ->  WT[N][K] bf16
// ---------------------------------------------------------------------------
__global__ __launch_bounds__(256) void transpose_bf16(
    const float* __restrict__ W, u16* __restrict__ WT, int K, int N) {
  __shared__ float t[32][33];
  int n0 = blockIdx.x * 32, k0 = blockIdx.y * 32;
#pragma unroll
  for (int i = 0; i < 4; ++i) {
    int r = threadIdx.y + i * 8;
    t[r][threadIdx.x] = W[(size_t)(k0 + r) * N + n0 + threadIdx.x];
  }
  __syncthreads();
#pragma unroll
  for (int i = 0; i < 4; ++i) {
    int r = threadIdx.y + i * 8;
    WT[(size_t)(n0 + r) * K + k0 + threadIdx.x] = f2bf(t[threadIdx.x][r]);
  }
}

// ---------------------------------------------------------------------------
// LayerNorm over C=1024, one wave per row, output bf16.
// ---------------------------------------------------------------------------
__global__ __launch_bounds__(256) void ln_bf16(
    const float* __restrict__ x, const float* __restrict__ g,
    const float* __restrict__ b, u16* __restrict__ out) {
  const int lane = threadIdx.x & 31, wid = threadIdx.x >> 5;
  const int row = blockIdx.x * 8 + wid;
  const float4* xr = (const float4*)(x + (size_t)row * 1024);
  float4 xv[8];
  float s = 0.f;
#pragma unroll
  for (int j = 0; j < 8; ++j) {
    xv[j] = xr[j * 32 + lane];
    s += xv[j].x + xv[j].y + xv[j].z + xv[j].w;
  }
#pragma unroll
  for (int off = 16; off; off >>= 1) s += __shfl_xor(s, off, 32);
  const float mu = s * (1.0f / 1024.0f);
  float v = 0.f;
#pragma unroll
  for (int j = 0; j < 8; ++j) {
    float a0 = xv[j].x - mu, a1 = xv[j].y - mu, a2 = xv[j].z - mu, a3 = xv[j].w - mu;
    v += a0 * a0 + a1 * a1 + a2 * a2 + a3 * a3;
  }
#pragma unroll
  for (int off = 16; off; off >>= 1) v += __shfl_xor(v, off, 32);
  const float rstd = rsqrtf(v * (1.0f / 1024.0f) + 1e-5f);
  uint2* orow = (uint2*)(out + (size_t)row * 1024);
  const float4* gr = (const float4*)g;
  const float4* br = (const float4*)b;
#pragma unroll
  for (int j = 0; j < 8; ++j) {
    float4 gv = gr[j * 32 + lane], bv = br[j * 32 + lane];
    float y0 = (xv[j].x - mu) * rstd * gv.x + bv.x;
    float y1 = (xv[j].y - mu) * rstd * gv.y + bv.y;
    float y2 = (xv[j].z - mu) * rstd * gv.z + bv.z;
    float y3 = (xv[j].w - mu) * rstd * gv.w + bv.w;
    uint2 p; p.x = pk2(y0, y1); p.y = pk2(y2, y3);
    orow[j * 32 + lane] = p;
  }
}

// ---------------------------------------------------------------------------
// GEMM: C[M,N] = A[M,K](bf16 row-major) x W (given as WT[N,K] bf16).
// Block 256 threads = 8 waves; block tile 128x128; wave tile 64x32 = 4x2 WMMA.
// Epilogues:
//   0: write bf16
//   1: write f32, += bias[n] + res[m*N+n]
//   2: write bf16, gelu(acc + bias[n])
//   3: write bf16 transposed into vT[bh][dh][t]  (V projection)
// ---------------------------------------------------------------------------
template <int EPI>
__global__ __launch_bounds__(256) void gemm_bf16(
    const u16* __restrict__ A, const u16* __restrict__ BT,
    int M, int N, int K, void* __restrict__ outp,
    const float* __restrict__ bias, const float* __restrict__ res) {
  const int lane = threadIdx.x & 31, wid = threadIdx.x >> 5;
  const int l15 = lane & 15, hi = lane >> 4;
  const int row0 = blockIdx.y * 128 + (wid >> 2) * 64;
  const int col0 = blockIdx.x * 128 + (wid & 3) * 32;

  v8f acc[4][2];
#pragma unroll
  for (int i = 0; i < 4; ++i)
#pragma unroll
    for (int j = 0; j < 2; ++j) acc[i][j] = v8f_zero();

  for (int k0 = 0; k0 < K; k0 += 32) {
    FragB af[4];
#pragma unroll
    for (int mt = 0; mt < 4; ++mt) {
      size_t p = (size_t)(row0 + mt * 16 + l15) * K + k0 + hi * 8;
      af[mt].q[0] = *(const uint4*)(A + p);
      af[mt].q[1] = *(const uint4*)(A + p + 16);
    }
    FragB bf[2];
#pragma unroll
    for (int nt = 0; nt < 2; ++nt) {
      size_t p = (size_t)(col0 + nt * 16 + l15) * K + k0 + hi * 8;
      bf[nt].q[0] = *(const uint4*)(BT + p);
      bf[nt].q[1] = *(const uint4*)(BT + p + 16);
    }
#pragma unroll
    for (int mt = 0; mt < 4; ++mt)
#pragma unroll
      for (int nt = 0; nt < 2; ++nt)
        acc[mt][nt] = wmma_bf(af[mt].v, bf[nt].v, acc[mt][nt]);
  }

  // Epilogue. D layout: lane<16 -> N=l15, M=v ; lane>=16 -> N=l15, M=v+8.
#pragma unroll
  for (int mt = 0; mt < 4; ++mt)
#pragma unroll
    for (int nt = 0; nt < 2; ++nt) {
      const int n = col0 + nt * 16 + l15;
      float bv = 0.f;
      if constexpr (EPI == 1 || EPI == 2) bv = bias[n];
#pragma unroll
      for (int v = 0; v < 8; ++v) {
        const int m = row0 + mt * 16 + hi * 8 + v;
        float val = acc[mt][nt][v];
        if constexpr (EPI == 0) {
          ((u16*)outp)[(size_t)m * N + n] = f2bf(val);
        } else if constexpr (EPI == 1) {
          size_t idx = (size_t)m * N + n;
          ((float*)outp)[idx] = val + bv + res[idx];
        } else if constexpr (EPI == 2) {
          float t = val + bv;
          float gl = 0.5f * t * (1.0f + erff(t * 0.70710678118654752f));
          ((u16*)outp)[(size_t)m * N + n] = f2bf(gl);
        } else {  // EPI == 3: vT[((b*16+head)*64+dh)*2048 + t]
          int bb = m >> 11, tt = m & 2047;
          int head = n >> 6, dh = n & 63;
          ((u16*)outp)[(((size_t)(bb * 16 + head) * 64 + dh) << 11) + tt] = f2bf(val);
        }
      }
    }
}

// ---------------------------------------------------------------------------
// Flash attention (causal). One block = 4 waves = 128 queries of one (b,h).
// Per wave: 32 queries; loop key blocks of 32.  Scores via QK^T WMMA,
// softmax row-pass through wave-private LDS (lane = query), then
// O^T += V^T x P^T WMMA (so output D-layout column == query == lane stats).
// ---------------------------------------------------------------------------
__global__ __launch_bounds__(128) void attn_kernel(
    const u16* __restrict__ qb, const u16* __restrict__ kb,
    const u16* __restrict__ vtb, u16* __restrict__ attb) {
  constexpr int T = 2048, C = 1024;
  __shared__ __align__(16) float Sl[4][32][36];   // scores, padded pitch
  __shared__ __align__(16) u16   Pl[4][32][40];   // probabilities bf16

  const int lane = threadIdx.x & 31, w = threadIdx.x >> 5;
  const int l15 = lane & 15, hi = lane >> 4, hi8 = hi * 8;
  const int bh = blockIdx.y, b = bh >> 4, h = bh & 15;
  const int qbase = blockIdx.x * 128 + w * 32;
  const size_t rowbase = ((size_t)b * T) * C + h * 64;

  // Q fragments: A-operand, 2 query tiles x 2 k-steps (dh 0..31 / 32..63)
  FragB qf[2][2];
#pragma unroll
  for (int mt = 0; mt < 2; ++mt)
#pragma unroll
    for (int ks = 0; ks < 2; ++ks) {
      size_t p = rowbase + (size_t)(qbase + mt * 16 + l15) * C + ks * 32 + hi * 8;
      qf[mt][ks].q[0] = *(const uint4*)(qb + p);
      qf[mt][ks].q[1] = *(const uint4*)(qb + p + 16);
    }

  v8f o[4][2];  // O^T tiles: rows = dh (4x16), cols = query (2x16)
#pragma unroll
  for (int i = 0; i < 4; ++i)
#pragma unroll
    for (int j = 0; j < 2; ++j) o[i][j] = v8f_zero();

  float mrow = -3.0e38f, lrow = 0.0f;  // per-lane = per-query stats
  const int qglob = qbase + lane;

  for (int k0 = 0; k0 <= qbase + 31; k0 += 32) {
    // ---- K fragments (B-operand of QK^T): lane = key column
    FragB kf[2][2];
#pragma unroll
    for (int nt = 0; nt < 2; ++nt)
#pragma unroll
      for (int ks = 0; ks < 2; ++ks) {
        size_t p = rowbase + (size_t)(k0 + nt * 16 + l15) * C + ks * 32 + hi * 8;
        kf[nt][ks].q[0] = *(const uint4*)(kb + p);
        kf[nt][ks].q[1] = *(const uint4*)(kb + p + 16);
      }
    // ---- S = Q K^T  (4 tiles), spill to LDS in [query][key] order
#pragma unroll
    for (int mt = 0; mt < 2; ++mt)
#pragma unroll
      for (int nt = 0; nt < 2; ++nt) {
        v8f s = wmma_bf(qf[mt][0].v, kf[nt][0].v, v8f_zero());
        s = wmma_bf(qf[mt][1].v, kf[nt][1].v, s);
#pragma unroll
        for (int v = 0; v < 8; ++v)
          Sl[w][mt * 16 + hi8 + v][nt * 16 + l15] = s[v];
      }
    __builtin_amdgcn_wave_barrier();  // DS ops are in-order within a wave

    // ---- row pass: lane owns query (qbase+lane); 32 scores contiguous
    float vals[32];
    const float4* rp = (const float4*)&Sl[w][lane][0];
#pragma unroll
    for (int j = 0; j < 8; ++j) {
      float4 t = rp[j];
      vals[4 * j + 0] = t.x; vals[4 * j + 1] = t.y;
      vals[4 * j + 2] = t.z; vals[4 * j + 3] = t.w;
    }
    float newm = mrow;
#pragma unroll
    for (int j = 0; j < 32; ++j) {
      float sc = vals[j] * 0.125f;                 // HEAD^-0.5 = 1/8
      if (k0 + j > qglob) sc = -3.0e38f;           // causal mask
      vals[j] = sc;
      newm = fmaxf(newm, sc);
    }
    const float alpha = __expf(mrow - newm);
    float ssum = 0.f;
#pragma unroll
    for (int j = 0; j < 32; ++j) {
      float p = __expf(vals[j] - newm);
      ssum += p;
      vals[j] = p;
    }
    lrow = lrow * alpha + ssum;
    mrow = newm;
    // write P row (bf16, contiguous) for the B-operand of V^T P^T
    uint4* pp = (uint4*)&Pl[w][lane][0];
#pragma unroll
    for (int j = 0; j < 4; ++j) {
      uint4 u;
      u.x = pk2(vals[8 * j + 0], vals[8 * j + 1]);
      u.y = pk2(vals[8 * j + 2], vals[8 * j + 3]);
      u.z = pk2(vals[8 * j + 4], vals[8 * j + 5]);
      u.w = pk2(vals[8 * j + 6], vals[8 * j + 7]);
      pp[j] = u;
    }
    __builtin_amdgcn_wave_barrier();

    // ---- rescale O^T columns by alpha (col = query = lane-held stat)
    const float a0 = __shfl(alpha, l15, 32);
    const float a1 = __shfl(alpha, 16 + l15, 32);
#pragma unroll
    for (int dt = 0; dt < 4; ++dt)
#pragma unroll
      for (int v = 0; v < 8; ++v) {
        o[dt][0][v] *= a0;
        o[dt][1][v] *= a1;
      }

    // ---- O^T += V^T x P^T : A from vT (contiguous), B from LDS P rows
    FragB vf[4];
#pragma unroll
    for (int dt = 0; dt < 4; ++dt) {
      size_t p = (((size_t)bh * 64) + dt * 16 + l15) * (size_t)T + k0 + hi * 8;
      vf[dt].q[0] = *(const uint4*)(vtb + p);
      vf[dt].q[1] = *(const uint4*)(vtb + p + 16);
    }
    FragB pf[2];
#pragma unroll
    for (int nt = 0; nt < 2; ++nt) {
      const uint4* pr = (const uint4*)&Pl[w][nt * 16 + l15][hi * 8];
      pf[nt].q[0] = pr[0];
      pf[nt].q[1] = pr[2];
    }
#pragma unroll
    for (int dt = 0; dt < 4; ++dt)
#pragma unroll
      for (int nt = 0; nt < 2; ++nt)
        o[dt][nt] = wmma_bf(vf[dt].v, pf[nt].v, o[dt][nt]);
    __builtin_amdgcn_wave_barrier();
  }

  // ---- finalize: divide columns by l, store att[t][h*64+dh] bf16 (packed b128)
  const float inv = 1.0f / lrow;
  const float i0 = __shfl(inv, l15, 32);
  const float i1 = __shfl(inv, 16 + l15, 32);
#pragma unroll
  for (int dt = 0; dt < 4; ++dt)
#pragma unroll
    for (int nt = 0; nt < 2; ++nt) {
      const float sc = nt ? i1 : i0;
      uint4 u;
      u.x = pk2(o[dt][nt][0] * sc, o[dt][nt][1] * sc);
      u.y = pk2(o[dt][nt][2] * sc, o[dt][nt][3] * sc);
      u.z = pk2(o[dt][nt][4] * sc, o[dt][nt][5] * sc);
      u.w = pk2(o[dt][nt][6] * sc, o[dt][nt][7] * sc);
      const int t = qbase + nt * 16 + l15;
      size_t p = ((size_t)b * T + t) * C + h * 64 + dt * 16 + hi8;
      *(uint4*)(attb + p) = u;
    }
}

// ---------------------------------------------------------------------------
// Host-side orchestration
// ---------------------------------------------------------------------------
extern "C" void kernel_launch(void* const* d_in, const int* in_sizes, int n_in,
                              void* d_out, int out_size, void* d_ws, size_t ws_size,
                              hipStream_t stream) {
  (void)in_sizes; (void)n_in; (void)out_size; (void)ws_size;
  const float* x   = (const float*)d_in[0];
  const float* Wq  = (const float*)d_in[1];
  const float* Wk  = (const float*)d_in[2];
  const float* Wv  = (const float*)d_in[3];
  const float* Wo  = (const float*)d_in[4];
  const float* bo  = (const float*)d_in[5];
  const float* W1  = (const float*)d_in[6];
  const float* b1  = (const float*)d_in[7];
  const float* W2  = (const float*)d_in[8];
  const float* b2  = (const float*)d_in[9];
  const float* g1  = (const float*)d_in[10];
  const float* be1 = (const float*)d_in[11];
  const float* g2  = (const float*)d_in[12];
  const float* be2 = (const float*)d_in[13];

  char* w = (char*)d_ws;
  auto alloc = [&](size_t bytes) -> char* {
    char* p = w;
    w += (bytes + 255) & ~(size_t)255;
    return p;
  };
  const size_t BT = 4096, Cc = 1024, FF = 4096;
  u16*   h1  = (u16*)alloc(BT * Cc * 2);
  u16*   WqT = (u16*)alloc(Cc * Cc * 2);
  u16*   WkT = (u16*)alloc(Cc * Cc * 2);
  u16*   WvT = (u16*)alloc(Cc * Cc * 2);
  u16*   WoT = (u16*)alloc(Cc * Cc * 2);
  u16*   W1T = (u16*)alloc(Cc * FF * 2);
  u16*   W2T = (u16*)alloc(FF * Cc * 2);
  u16*   qbf = (u16*)alloc(BT * Cc * 2);
  u16*   kbf = (u16*)alloc(BT * Cc * 2);
  u16*   vT  = (u16*)alloc(BT * Cc * 2);
  u16*   att = (u16*)alloc(BT * Cc * 2);
  float* x2  = (float*)alloc(BT * Cc * 4);
  u16*   h2  = (u16*)alloc(BT * Cc * 2);
  u16*   f1  = (u16*)alloc(BT * FF * 2);

  dim3 tb(32, 8);
  transpose_bf16<<<dim3(32, 32),  tb, 0, stream>>>(Wq, WqT, 1024, 1024);
  transpose_bf16<<<dim3(32, 32),  tb, 0, stream>>>(Wk, WkT, 1024, 1024);
  transpose_bf16<<<dim3(32, 32),  tb, 0, stream>>>(Wv, WvT, 1024, 1024);
  transpose_bf16<<<dim3(32, 32),  tb, 0, stream>>>(Wo, WoT, 1024, 1024);
  transpose_bf16<<<dim3(128, 32), tb, 0, stream>>>(W1, W1T, 1024, 4096);
  transpose_bf16<<<dim3(32, 128), tb, 0, stream>>>(W2, W2T, 4096, 1024);

  ln_bf16<<<512, 256, 0, stream>>>(x, g1, be1, h1);

  gemm_bf16<0><<<dim3(8, 32), 256, 0, stream>>>(h1, WqT, 4096, 1024, 1024, qbf, nullptr, nullptr);
  gemm_bf16<0><<<dim3(8, 32), 256, 0, stream>>>(h1, WkT, 4096, 1024, 1024, kbf, nullptr, nullptr);
  gemm_bf16<3><<<dim3(8, 32), 256, 0, stream>>>(h1, WvT, 4096, 1024, 1024, vT,  nullptr, nullptr);

  attn_kernel<<<dim3(16, 32), 128, 0, stream>>>(qbf, kbf, vT, att);

  gemm_bf16<1><<<dim3(8, 32), 256, 0, stream>>>(att, WoT, 4096, 1024, 1024, x2, bo, x);

  ln_bf16<<<512, 256, 0, stream>>>(x2, g2, be2, h2);

  gemm_bf16<2><<<dim3(32, 32), 256, 0, stream>>>(h2, W1T, 4096, 4096, 1024, f1, b1, nullptr);
  gemm_bf16<1><<<dim3(8, 32),  256, 0, stream>>>(f1, W2T, 4096, 1024, 4096, (float*)d_out, b2, x2);
}
